// LocAtt_20229295964633
// MI455X (gfx1250) — matrix operand order repaired and checked by reading, not compiled, not run on hardware
//
#include <hip/hip_runtime.h>
#include <hip/hip_bf16.h>

typedef __attribute__((ext_vector_type(16))) _Float16 v16h;
typedef __attribute__((ext_vector_type(8)))  _Float16 v8h;
typedef __attribute__((ext_vector_type(4)))  _Float16 v4h;
typedef __attribute__((ext_vector_type(8)))  float    v8f;
typedef __attribute__((ext_vector_type(4)))  int      v4i;

#define C_DIM 128
#define HW    4096   // 64*64
#define NB    8      // batch
#define NP    49     // 7x7 neighborhood
#define TS    8      // stage-2 spatial tile side
#define HALO  14     // TS + 6

// Async global->LDS copy path (CDNA5 GLOBAL_LOAD_ASYNC_TO_LDS_B128, ASYNCcnt).
// Round-3 diagnostics confirmed the builtin exists, takes 4 args, and wants
// a generic 'int4*' as the first (global source) parameter.
#if __has_builtin(__builtin_amdgcn_global_load_async_to_lds_b128) && \
    __has_builtin(__builtin_amdgcn_s_wait_asynccnt)
#define USE_ASYNC_LDS 1
#else
#define USE_ASYNC_LDS 0
#endif

typedef __attribute__((address_space(3))) v4i* lds_v4i_ptr;

// ---------------------------------------------------------------------------
// Kernel 0: convert Wq|Wk|Wv (each 128x128 fp32) to fp16, packed contiguously.
// ---------------------------------------------------------------------------
__global__ void wconvert_kernel(const float* __restrict__ Wq,
                                const float* __restrict__ Wk,
                                const float* __restrict__ Wv,
                                _Float16* __restrict__ w16) {
  int i = blockIdx.x * 256 + threadIdx.x;          // 0 .. 49151
  if (i < 16384)        w16[i] = (_Float16)Wq[i];
  else if (i < 32768)   w16[i] = (_Float16)Wk[i - 16384];
  else                  w16[i] = (_Float16)Wv[i - 32768];
}

// ---------------------------------------------------------------------------
// Kernel 1: QKV projection GEMM with WMMA (unchanged; 96 v_wmma per wave,
// s_clause'd global b128 A-loads, LDS b128 B-loads, split wait counters).
//   out[o, n] = sum_c W[o,c] * x[b,c,n] + bias[o]    (M=128, K=128, N=4096/b)
// A-fragment (16x32 f16, ISA 7.12.2): lane l row = 16w + (l&15);
//   lanes 0-15 hold K {0..7, 16..23}, lanes 16-31 hold K {8..15, 24..31}.
// B-fragment (32x16 f16): lane l column N = l&15;
//   lanes 0-15 hold K {0..7, 16..23}, lanes 16-31 hold K {8..15, 24..31} (2 packed per VGPR).
// C/D (16x16 f32): VGPR r = row M=r (lanes 0-15) / M=8+r (lanes 16-31).
// Output q16/k16/v16 stored channel-last fp16: [b][pixel][c].
// ---------------------------------------------------------------------------
__global__ __launch_bounds__(256) void qkv_gemm_kernel(
    const float*   __restrict__ x,
    const _Float16* __restrict__ w16,
    const float* __restrict__ bq, const float* __restrict__ bk,
    const float* __restrict__ bv,
    _Float16* __restrict__ q16, _Float16* __restrict__ k16,
    _Float16* __restrict__ v16) {
  __shared__ _Float16 sX[128][136];                // 34 KB, +8 f16 row pad

  const int b    = blockIdx.x >> 5;                // 8 batches
  const int pix0 = (blockIdx.x & 31) << 7;         // 32 tiles of 128 pixels
  const int tid  = threadIdx.x;

  for (int i = tid; i < 128 * 128; i += 256) {
    int c = i >> 7, p = i & 127;
    sX[p][c] = (_Float16)x[((size_t)(b * C_DIM + c) << 12) + pix0 + p];
  }
  __syncthreads();

  const int wave = tid >> 5;
  const int lane = tid & 31;
  const int mo   = wave << 4;
  const int hs   = (lane < 16) ? 0 : 1;
  const int n15  = lane & 15;

  const float*    biases[3] = {bq, bk, bv};
  _Float16*       outs[3]   = {q16, k16, v16};

  #pragma unroll
  for (int mat = 0; mat < 3; ++mat) {
    const _Float16* Wrow = w16 + mat * 16384 + (mo + n15) * C_DIM;
    v16h afrag[4];
    #pragma unroll
    for (int kc = 0; kc < 4; ++kc) {
      const _Float16* ap = Wrow + kc * 32 + hs * 8;
      v8h lo = *(const v8h*)ap;
      v8h hi = *(const v8h*)(ap + 16);
      #pragma unroll
      for (int e = 0; e < 8; ++e) { afrag[kc][e] = lo[e]; afrag[kc][e + 8] = hi[e]; }
    }
    const float* bias = biases[mat];
    float bcol[8];
    #pragma unroll
    for (int r = 0; r < 8; ++r) bcol[r] = bias[mo + hs * 8 + r];

    _Float16* outp = outs[mat] + ((size_t)b * HW + pix0) * C_DIM;

    for (int nt = 0; nt < 8; ++nt) {
      v8f acc = {};
      #pragma unroll
      for (int kc = 0; kc < 4; ++kc) {
        const _Float16* bp = &sX[nt * 16 + n15][kc * 32 + hs * 8];
        v8h blo = *(const v8h*)bp;
        v8h bhi = *(const v8h*)(bp + 16);
        v16h bfrag;
        #pragma unroll
        for (int e = 0; e < 8; ++e) { bfrag[e] = blo[e]; bfrag[e + 8] = bhi[e]; }
        acc = __builtin_amdgcn_wmma_f32_16x16x32_f16(
                  false, afrag[kc], false, bfrag,
                  (short)0, acc, false, false);
      }
      v8h o8;
      #pragma unroll
      for (int r = 0; r < 8; ++r) o8[r] = (_Float16)(acc[r] + bcol[r]);
      *(v8h*)(outp + (size_t)(nt * 16 + n15) * C_DIM + mo + hs * 8) = o8;
    }
  }
}

// ---------------------------------------------------------------------------
// Kernel 2: local 7x7 attention with LDS-staged k/v halo (98 KB dynamic LDS).
// Block = 256 threads (8 waves) owns an 8x8 pixel tile; the 14x14 halo of
// k and v is staged to LDS — via ASYNC global->LDS copies where available —
// with OOB pixels zero-filled (== reference zero-padding: OOB logit 0 inside
// the softmax, OOB v = 0; no bounds checks in the inner loops).
// ---------------------------------------------------------------------------
__global__ __launch_bounds__(256) void locatt_kernel(
    const _Float16* __restrict__ q16, const _Float16* __restrict__ k16,
    const _Float16* __restrict__ v16, float* __restrict__ out) {
  extern __shared__ _Float16 smem[];
  _Float16* sK = smem;                          // HALO*HALO*128 f16
  _Float16* sV = smem + HALO * HALO * C_DIM;

  const int tid = threadIdx.x;
  const int b   = blockIdx.x >> 6;              // 8 images
  const int th  = (blockIdx.x >> 3) & 7;        // tile row
  const int tw  = blockIdx.x & 7;               // tile col
  const int h0  = th * TS, w0 = tw * TS;

  // ---- stage k/v halo: 196 pixels x 16 chunks of 16 B per matrix ----------
  for (int id = tid; id < HALO * HALO * 16; id += 256) {
    const int pxl  = id >> 4;                   // halo pixel 0..195
    const int part = id & 15;                   // 16-B chunk within 256-B line
    const int r  = pxl / HALO;
    const int cc = pxl - r * HALO;
    const int gh = h0 - 3 + r, gw = w0 - 3 + cc;
    const int dso = pxl * C_DIM + part * 8;     // f16 offset in LDS
    if ((unsigned)gh < 64u && (unsigned)gw < 64u) {
      const size_t goff = ((size_t)((b << 12) + (gh << 6) + gw) << 7) + part * 8;
#if USE_ASYNC_LDS
      // Direct global->LDS DMA, no VGPR round-trip; tracked by ASYNCcnt.
      // src: generic int4* (per round-3 diagnostic); dst: AS(3) int4* built
      // from the flat pointer's low 32 bits (the LDS byte offset).
      __builtin_amdgcn_global_load_async_to_lds_b128(
          (v4i*)(k16 + goff),
          (lds_v4i_ptr)(unsigned)(uintptr_t)&sK[dso], 0, 0);
      __builtin_amdgcn_global_load_async_to_lds_b128(
          (v4i*)(v16 + goff),
          (lds_v4i_ptr)(unsigned)(uintptr_t)&sV[dso], 0, 0);
#else
      *(v8h*)&sK[dso] = *(const v8h*)(k16 + goff);
      *(v8h*)&sV[dso] = *(const v8h*)(v16 + goff);
#endif
    } else {
      v8h z = {};
      *(v8h*)&sK[dso] = z;                      // zero pad == reference pad
      *(v8h*)&sV[dso] = z;
    }
  }
#if USE_ASYNC_LDS
  __builtin_amdgcn_s_wait_asynccnt(0);          // our async copies done
#endif
  __syncthreads();                              // all waves' copies visible

  const int wave = tid >> 5;                    // tile row owned by this wave
  const int lane = tid & 31;                    // 4 channels per lane

  for (int pc = 0; pc < TS; ++pc) {             // 8 pixels along the row
    const int h = h0 + wave, w = w0 + pc;
    const int n = (h << 6) + w;

    const _Float16* qp = q16 + ((size_t)((b << 12) + n) << 7) + lane * 4;
    v4h qv = *(const v4h*)qp;                   // coalesced 8 B / lane
    const float q0 = (float)qv[0], q1 = (float)qv[1],
                q2 = (float)qv[2], q3 = (float)qv[3];

    float scores[NP];
    #pragma unroll
    for (int p = 0; p < NP; ++p) {
      const int hr = wave + p / 7;              // halo coords of neighbor
      const int hc = pc   + p % 7;
      const _Float16* kp = &sK[(hr * HALO + hc) * C_DIM + lane * 4];
      v4h kv = *(const v4h*)kp;                 // ds_load_b64, conflict-free
      float part = q0 * (float)kv[0] + q1 * (float)kv[1] +
                   q2 * (float)kv[2] + q3 * (float)kv[3];
      part += __shfl_xor(part, 16);
      part += __shfl_xor(part, 8);
      part += __shfl_xor(part, 4);
      part += __shfl_xor(part, 2);
      part += __shfl_xor(part, 1);
      scores[p] = part;                         // wave-uniform (0 if OOB)
    }

    float m = scores[0];
    #pragma unroll
    for (int p = 1; p < NP; ++p) m = fmaxf(m, scores[p]);

    float denom = 0.0f, y0 = 0.0f, y1 = 0.0f, y2 = 0.0f, y3 = 0.0f;
    #pragma unroll
    for (int p = 0; p < NP; ++p) {
      const float e = __expf(scores[p] - m);
      denom += e;
      const int hr = wave + p / 7;
      const int hc = pc   + p % 7;
      const _Float16* vp = &sV[(hr * HALO + hc) * C_DIM + lane * 4];
      v4h vv = *(const v4h*)vp;                 // zeros where OOB
      y0 += e * (float)vv[0]; y1 += e * (float)vv[1];
      y2 += e * (float)vv[2]; y3 += e * (float)vv[3];
    }
    const float inv = 1.0f / denom;

    // y layout (B, C, H, W): channel stride = 4096 floats
    const int c = lane * 4;
    float* op = out + ((size_t)b * C_DIM + c) * HW + n;
    op[0]     = y0 * inv;
    op[4096]  = y1 * inv;
    op[8192]  = y2 * inv;
    op[12288] = y3 * inv;
  }
}

// ---------------------------------------------------------------------------
// Workspace layout (bytes):
//   [0, 98304)                 fp16 weights Wq|Wk|Wv
//   [98304, +8 MB)             q16  (B*HW*C fp16, channel-last)
//   [.., +8 MB)                k16
//   [.., +8 MB)                v16          total ~25.3 MB
// ---------------------------------------------------------------------------
extern "C" void kernel_launch(void* const* d_in, const int* in_sizes, int n_in,
                              void* d_out, int out_size, void* d_ws, size_t ws_size,
                              hipStream_t stream) {
  const float* x  = (const float*)d_in[0];
  const float* Wq = (const float*)d_in[1];
  const float* bq = (const float*)d_in[2];
  const float* Wk = (const float*)d_in[3];
  const float* bk = (const float*)d_in[4];
  const float* Wv = (const float*)d_in[5];
  const float* bv = (const float*)d_in[6];

  char* ws = (char*)d_ws;
  const size_t qkv_bytes = (size_t)NB * HW * C_DIM * sizeof(_Float16); // 8 MB
  _Float16* w16 = (_Float16*)ws;
  _Float16* q16 = (_Float16*)(ws + 98304);
  _Float16* k16 = (_Float16*)(ws + 98304 + qkv_bytes);
  _Float16* v16 = (_Float16*)(ws + 98304 + 2 * qkv_bytes);

  wconvert_kernel<<<192, 256, 0, stream>>>(Wq, Wk, Wv, w16);
  qkv_gemm_kernel<<<NB * 32, 256, 0, stream>>>(x, w16, bq, bk, bv, q16, k16, v16);

  const size_t smem2 = (size_t)2 * HALO * HALO * C_DIM * sizeof(_Float16); // 100352 B
  locatt_kernel<<<NB * (64 / TS) * (64 / TS), 256, smem2, stream>>>(
      q16, k16, v16, (float*)d_out);
}
